// SwinBlock_64965675319859
// MI455X (gfx1250) — matrix operand order repaired and checked by reading, not compiled, not run on hardware
//
#include <hip/hip_runtime.h>
#include <math.h>

// ---------------- static problem config (matches reference) ----------------
// B=8, H=32, W=512, C=256, NH=8, window 8x8 (N=64), shift 4, HID=1024
#define CB   8
#define CH   32
#define CW   512
#define CC   256
#define CNH  8
#define CN   64
#define CNW  256            // (32/8)*(512/8)
#define CHID 1024

typedef __attribute__((ext_vector_type(16))) __bf16 bf16x16;
typedef __attribute__((ext_vector_type(8)))  float  f32x8;

// ---------------- bf16 helpers ----------------
__device__ __forceinline__ unsigned short f2bf(float f) {
    unsigned u = __float_as_uint(f);
    u += 0x7fffu + ((u >> 16) & 1u);     // round-to-nearest-even
    return (unsigned short)(u >> 16);
}
__device__ __forceinline__ float bf2f(unsigned short h) {
    return __uint_as_float(((unsigned)h) << 16);
}
__device__ __forceinline__ __bf16 us2bf(unsigned short h) {
    return __builtin_bit_cast(__bf16, h);
}

// ---------------- WMMA wrapper ----------------
__device__ __forceinline__ f32x8 wmma_bf16(bf16x16 a, bf16x16 b, f32x8 c) {
    return __builtin_amdgcn_wmma_f32_16x16x32_bf16(
        /*neg_a=*/false, a, /*neg_b=*/false, b,
        /*c_mod=*/(short)0, c, /*reuse_a=*/false, /*reuse_b=*/false);
}

// A fragment: 16x32 bf16 tile from row-major [M][ld]; p0 -> (row0, k0)
// lanes 0-15: rows, K in {0..7,16..23}; lanes 16-31: K offset +8 (ISA 7.12.2)
__device__ __forceinline__ bf16x16 load_frag_A(const unsigned short* p0, int ld) {
    const int lane = threadIdx.x & 31;
    const unsigned short* p = p0 + (lane & 15) * ld;
    const int kg = (lane >> 4) << 3;
    bf16x16 a;
#pragma unroll
    for (int v = 0; v < 8; ++v) {
        int kb = ((v < 4) ? 0 : 16) + kg + ((v & 3) << 1);
        a[2 * v]     = us2bf(p[kb]);
        a[2 * v + 1] = us2bf(p[kb + 1]);
    }
    return a;
}

// B fragment from weight-style [N][K] storage (B[k][n] = W[n][k]); p0 -> (n0,k0)
// lanes 0-15: K in {0..7,16..23}; lanes 16-31: K offset +8 (same as A)
__device__ __forceinline__ bf16x16 load_frag_B_nk(const unsigned short* p0, int ld) {
    const int lane = threadIdx.x & 31;
    const unsigned short* p = p0 + (lane & 15) * ld + ((lane >> 4) << 3);
    bf16x16 b;
#pragma unroll
    for (int v = 0; v < 8; ++v) {
        int kb = ((v < 4) ? 0 : 16) + ((v & 3) << 1);
        b[2 * v]     = us2bf(p[kb]);
        b[2 * v + 1] = us2bf(p[kb + 1]);
    }
    return b;
}

// ---------------- prep kernels ----------------
__global__ void conv_bf16_kernel(const float* __restrict__ src,
                                 unsigned short* __restrict__ dst, int n) {
    int i = blockIdx.x * 256 + threadIdx.x;
    if (i < n) dst[i] = f2bf(src[i]);
}

// continuous relative position bias MLP: 225 table entries -> [225][NH]
__global__ void cpb_kernel(const float* __restrict__ w1, const float* __restrict__ b1,
                           const float* __restrict__ w2, float* __restrict__ rpb) {
    const int i = blockIdx.x;          // 0..224
    const int t = threadIdx.x;         // 64 threads
    const int dh = i / 15 - 7, dw = i % 15 - 7;
    auto tf = [](int d) {
        float v = (float)d * (8.0f / 7.0f);
        float s = (v > 0.f) ? 1.f : ((v < 0.f) ? -1.f : 0.f);
        return s * log2f(fabsf(v) + 1.f) * (1.f / 3.f);  // /log2(8)
    };
    const float c0 = tf(dh), c1 = tf(dw);
    float ph[8];
#pragma unroll
    for (int h = 0; h < 8; ++h) ph[h] = 0.f;
    for (int j = t; j < 512; j += 64) {
        float a = fmaxf(w1[j * 2] * c0 + w1[j * 2 + 1] * c1 + b1[j], 0.f);
#pragma unroll
        for (int h = 0; h < 8; ++h) ph[h] += a * w2[h * 512 + j];
    }
    __shared__ float red[64][8];
#pragma unroll
    for (int h = 0; h < 8; ++h) red[t][h] = ph[h];
    __syncthreads();
    if (t < 8) {
        float s = 0.f;
        for (int k = 0; k < 64; ++k) s += red[k][t];
        rpb[i * 8 + t] = 16.f / (1.f + __expf(-s));
    }
}

// ---------------- main per-window kernel ----------------
// LDS pool layout (bytes) -- exactly 160KB so 2 workgroups fit per 320KB WGP:
#define OFF_Q    0        // bf16 q[8][64][32]        32KB
#define OFF_K    32768    // bf16 k[8][64][32]        32KB
#define OFF_V    65536    // bf16 vT[8][32][64]       32KB (d-major per head)
#define OFF_LN   98304    // bf16 ln [64][256]        32KB  (dead before P)
#define OFF_P    98304    // bf16 P[8][64][64]        64KB
#define OFF_O    0        // bf16 O [64][256]         32KB  (reuses Q)
#define OFF_X1   32768    // f32  x1[64][256]         64KB  (reuses K,V)
#define OFF_LN2  98304    // bf16 ln2[64][256]        32KB  (reuses P)
#define OFF_M1   131072   // bf16 m1 chunk [64][256]  32KB
#define POOL_BYTES 163840

__global__ __launch_bounds__(256)
void swin_window_kernel(const float* __restrict__ x,
                        const float* __restrict__ n1g, const float* __restrict__ n1b,
                        const float* __restrict__ q_bias, const float* __restrict__ v_bias,
                        const float* __restrict__ logit_scale,
                        const float* __restrict__ b_proj,
                        const float* __restrict__ n2g, const float* __restrict__ n2b,
                        const float* __restrict__ b_fc1, const float* __restrict__ b_fc2,
                        const unsigned short* __restrict__ wqkv,
                        const unsigned short* __restrict__ wproj,
                        const unsigned short* __restrict__ wfc1,
                        const unsigned short* __restrict__ wfc2,
                        const float* __restrict__ rpb,
                        float* __restrict__ out) {
    extern __shared__ char pool[];
    unsigned short* sQ   = (unsigned short*)(pool + OFF_Q);
    unsigned short* sK   = (unsigned short*)(pool + OFF_K);
    unsigned short* sV   = (unsigned short*)(pool + OFF_V);   // transposed: [h][d][tok]
    unsigned short* sLN  = (unsigned short*)(pool + OFF_LN);
    unsigned short* sP   = (unsigned short*)(pool + OFF_P);
    unsigned short* sO   = (unsigned short*)(pool + OFF_O);
    float*          sX1  = (float*)(pool + OFF_X1);
    unsigned short* sLN2 = (unsigned short*)(pool + OFF_LN2);
    unsigned short* sM1  = (unsigned short*)(pool + OFF_M1);

    const int wg   = blockIdx.x;          // window id: 0..2047
    const int b    = wg / CNW;
    const int wid  = wg - b * CNW;
    const int wi   = wid >> 6;            // window row 0..3
    const int wj   = wid & 63;            // window col 0..63
    const int tid  = threadIdx.x;
    const int lane = tid & 31;
    const int wv   = tid >> 5;            // wave 0..7

    // token t (0..63), channel c -> flat index into original x (fold shift+partition)
    auto gidx = [&](int t, int c) {
        int hs = wi * 8 + (t >> 3);
        int ws = wj * 8 + (t & 7);
        int oh = (hs + 4) & 31;
        int ow = (ws + 4) & 511;
        return ((b * CH + oh) * CW + ow) * CC + c;
    };
    // shift-mask region of token t
    auto region = [&](int t) {
        int hs = wi * 8 + (t >> 3);
        int ws = wj * 8 + (t & 7);
        int hb = (hs < 24) ? 0 : ((hs < 28) ? 1 : 2);
        int wb = (ws < 504) ? 0 : ((ws < 508) ? 1 : 2);
        return hb * 3 + wb;
    };

    // ---------------- Phase 1: LayerNorm1 -> sLN (bf16) ----------------
    for (int t = wv * 8; t < wv * 8 + 8; ++t) {
        float v[8];
        float s = 0.f;
#pragma unroll
        for (int i = 0; i < 8; ++i) { v[i] = x[gidx(t, lane + i * 32)]; s += v[i]; }
#pragma unroll
        for (int m = 16; m >= 1; m >>= 1) s += __shfl_xor(s, m, 32);
        float mu = s * (1.f / 256.f);
        float vs = 0.f;
#pragma unroll
        for (int i = 0; i < 8; ++i) { float d = v[i] - mu; vs += d * d; }
#pragma unroll
        for (int m = 16; m >= 1; m >>= 1) vs += __shfl_xor(vs, m, 32);
        float rstd = rsqrtf(vs * (1.f / 256.f) + 1e-5f);
#pragma unroll
        for (int i = 0; i < 8; ++i) {
            int c = lane + i * 32;
            sLN[t * 256 + c] = f2bf((v[i] - mu) * rstd * n1g[c] + n1b[c]);
        }
    }
    __syncthreads();

    // ---------------- Phase 2: QKV GEMM (64x768x256) ----------------
    for (int tile = wv; tile < 192; tile += 8) {
        int tm = tile & 3, tn = tile >> 2;
        f32x8 acc = {0, 0, 0, 0, 0, 0, 0, 0};
        for (int k0 = 0; k0 < 256; k0 += 32) {
            __builtin_prefetch(wqkv + (tn * 16) * 256 + k0 + 64, 0, 3);
            bf16x16 a = load_frag_A(sLN + tm * 16 * 256 + k0, 256);
            bf16x16 bb = load_frag_B_nk(wqkv + (tn * 16) * 256 + k0, 256);
            acc = wmma_bf16(a, bb, acc);
        }
        int n0 = tn * 16 + (lane & 15);
        int cc = n0 & 255;
        int head = cc >> 5, d = cc & 31;
        int m0 = tm * 16 + ((lane >> 4) << 3);
        if (n0 < 512) {
            float bias = (n0 < 256) ? q_bias[n0] : 0.f;
            unsigned short* dst = (n0 < 256) ? sQ : sK;
#pragma unroll
            for (int r = 0; r < 8; ++r)
                dst[(head * 64 + m0 + r) * 32 + d] = f2bf(acc[r] + bias);
        } else {
            float bias = v_bias[n0 - 512];
#pragma unroll
            for (int r = 0; r < 8; ++r)     // transposed store: [h][d][tok]
                sV[(head * 32 + d) * 64 + (m0 + r)] = f2bf(acc[r] + bias);
        }
    }
    __syncthreads();

    // ---------------- Phase 3: normalize q (x scale) and k ----------------
    {
        const float lmax = 4.6051701859880914f;  // ln(100)
        for (int rr = tid; rr < 512; rr += 256) {
            int h = rr >> 6;
            float sc = __expf(fminf(logit_scale[h], lmax));
            unsigned short* p = sQ + rr * 32;
            float ss = 0.f;
#pragma unroll
            for (int d = 0; d < 32; ++d) { float f = bf2f(p[d]); ss += f * f; }
            float inv = sc / fmaxf(sqrtf(ss), 1e-12f);
#pragma unroll
            for (int d = 0; d < 32; ++d) p[d] = f2bf(bf2f(p[d]) * inv);
        }
        for (int rr = tid; rr < 512; rr += 256) {
            unsigned short* p = sK + rr * 32;
            float ss = 0.f;
#pragma unroll
            for (int d = 0; d < 32; ++d) { float f = bf2f(p[d]); ss += f * f; }
            float inv = 1.f / fmaxf(sqrtf(ss), 1e-12f);
#pragma unroll
            for (int d = 0; d < 32; ++d) p[d] = f2bf(bf2f(p[d]) * inv);
        }
    }
    __syncthreads();

    // ------- Phase 4: attention scores + softmax + O = P@V (wave = head) ---
    {
        const int h = wv;
        const unsigned short* qh = sQ + h * 64 * 32;
        const unsigned short* kh = sK + h * 64 * 32;
        const unsigned short* vh = sV + h * 32 * 64;  // [d][tok]
        unsigned short* ph = sP + h * 64 * 64;
        const int g = lane >> 4, nl = lane & 15;

        // per-lane column metadata, hoisted out of the softmax loop
        int regj[4], rj4[4], cj4[4];
#pragma unroll
        for (int tn = 0; tn < 4; ++tn) {
            int n = tn * 16 + nl;
            regj[tn] = region(n);
            rj4[tn] = n >> 3;
            cj4[tn] = n & 7;
        }

        float rsum[4][8];   // 1/rowsum, indexed exactly as O tiles consume it
#pragma unroll
        for (int tm = 0; tm < 4; ++tm) {
            // one 16-row block: 4 live accumulator tiles (32 VGPRs)
            bf16x16 a = load_frag_A(qh + tm * 16 * 32, 32);
            f32x8 S[4];
#pragma unroll
            for (int tn = 0; tn < 4; ++tn) {
                f32x8 z = {0, 0, 0, 0, 0, 0, 0, 0};
                bf16x16 bb = load_frag_B_nk(kh + tn * 16 * 32, 32);  // B[d][tok]=K[tok][d]
                S[tn] = wmma_bf16(a, bb, z);
            }
#pragma unroll
            for (int r = 0; r < 8; ++r) {
                int m = tm * 16 + g * 8 + r;
                int rm = region(m);
                int ri = m >> 3, ci = m & 7;
                float vv[4];
                float vmax = -1e30f;
#pragma unroll
                for (int tn = 0; tn < 4; ++tn) {
                    int idx = (ri - rj4[tn] + 7) * 15 + (ci - cj4[tn] + 7);
                    float val = S[tn][r] + rpb[idx * 8 + h];
                    if (regj[tn] != rm) val -= 100.f;
                    vv[tn] = val;
                    vmax = fmaxf(vmax, val);
                }
#pragma unroll
                for (int mm = 8; mm >= 1; mm >>= 1)
                    vmax = fmaxf(vmax, __shfl_xor(vmax, mm, 32));
                float ssum = 0.f;
#pragma unroll
                for (int tn = 0; tn < 4; ++tn) {
                    float e = __expf(vv[tn] - vmax);
                    ssum += e;
                    ph[m * 64 + tn * 16 + nl] = f2bf(e);
                }
#pragma unroll
                for (int mm = 8; mm >= 1; mm >>= 1) ssum += __shfl_xor(ssum, mm, 32);
                rsum[tm][r] = 1.f / ssum;   // all lanes in group g hold row m's sum
            }
        }
        __syncthreads();  // q,k fully consumed; sO may overwrite sQ region

        // O = P @ V, normalized by row sums held in registers
#pragma unroll
        for (int tm = 0; tm < 4; ++tm)
#pragma unroll
            for (int tn = 0; tn < 2; ++tn) {
                f32x8 acc = {0, 0, 0, 0, 0, 0, 0, 0};
#pragma unroll
                for (int k0 = 0; k0 < 64; k0 += 32) {
                    bf16x16 a = load_frag_A(ph + tm * 16 * 64 + k0, 64);
                    bf16x16 bb = load_frag_B_nk(vh + (tn * 16) * 64 + k0, 64);
                    acc = wmma_bf16(a, bb, acc);
                }
                int m0 = tm * 16 + (g << 3);
                int n = tn * 16 + nl;
#pragma unroll
                for (int r = 0; r < 8; ++r)
                    sO[(m0 + r) * 256 + h * 32 + n] = f2bf(acc[r] * rsum[tm][r]);
            }
    }
    __syncthreads();

    // ---------------- Phase 5: proj + residual -> sX1 (fp32) ----------------
    for (int tile = wv; tile < 64; tile += 8) {
        int tm = tile & 3, tn = tile >> 2;
        f32x8 acc = {0, 0, 0, 0, 0, 0, 0, 0};
        for (int k0 = 0; k0 < 256; k0 += 32) {
            __builtin_prefetch(wproj + (tn * 16) * 256 + k0 + 64, 0, 3);
            bf16x16 a = load_frag_A(sO + tm * 16 * 256 + k0, 256);
            bf16x16 bb = load_frag_B_nk(wproj + (tn * 16) * 256 + k0, 256);
            acc = wmma_bf16(a, bb, acc);
        }
        int n = tn * 16 + (lane & 15);
        int m0 = tm * 16 + ((lane >> 4) << 3);
        float bp = b_proj[n];
#pragma unroll
        for (int r = 0; r < 8; ++r) {
            int m = m0 + r;
            sX1[m * 256 + n] = x[gidx(m, n)] + acc[r] + bp;
        }
    }
    __syncthreads();

    // ---------------- Phase 6: LayerNorm2 -> sLN2 (bf16) ----------------
    for (int t = wv * 8; t < wv * 8 + 8; ++t) {
        float v[8];
        float s = 0.f;
#pragma unroll
        for (int i = 0; i < 8; ++i) { v[i] = sX1[t * 256 + lane + i * 32]; s += v[i]; }
#pragma unroll
        for (int m = 16; m >= 1; m >>= 1) s += __shfl_xor(s, m, 32);
        float mu = s * (1.f / 256.f);
        float vs = 0.f;
#pragma unroll
        for (int i = 0; i < 8; ++i) { float d = v[i] - mu; vs += d * d; }
#pragma unroll
        for (int m = 16; m >= 1; m >>= 1) vs += __shfl_xor(vs, m, 32);
        float rstd = rsqrtf(vs * (1.f / 256.f) + 1e-5f);
#pragma unroll
        for (int i = 0; i < 8; ++i) {
            int c = lane + i * 32;
            sLN2[t * 256 + c] = f2bf((v[i] - mu) * rstd * n2g[c] + n2b[c]);
        }
    }
    __syncthreads();

    // ---------------- Phase 7: MLP (4 hidden chunks of 256) ----------------
    {
        f32x8 acc2[8];
#pragma unroll
        for (int i = 0; i < 8; ++i) acc2[i] = (f32x8){0, 0, 0, 0, 0, 0, 0, 0};
        for (int cc4 = 0; cc4 < 4; ++cc4) {
            // m1 chunk = gelu(ln2 @ W1c^T + b1c) -> sM1 (bf16)
            for (int tile = wv; tile < 64; tile += 8) {
                int tm = tile & 3, tn = tile >> 2;
                f32x8 acc = {0, 0, 0, 0, 0, 0, 0, 0};
                for (int k0 = 0; k0 < 256; k0 += 32) {
                    __builtin_prefetch(wfc1 + (cc4 * 256 + tn * 16) * 256 + k0 + 64, 0, 3);
                    bf16x16 a = load_frag_A(sLN2 + tm * 16 * 256 + k0, 256);
                    bf16x16 bb = load_frag_B_nk(wfc1 + (cc4 * 256 + tn * 16) * 256 + k0, 256);
                    acc = wmma_bf16(a, bb, acc);
                }
                int n = tn * 16 + (lane & 15);
                int ng = cc4 * 256 + n;
                int m0 = tm * 16 + ((lane >> 4) << 3);
                float b1v = b_fc1[ng];
#pragma unroll
                for (int r = 0; r < 8; ++r) {
                    float u = acc[r] + b1v;
                    float gel = 0.5f * u * (1.f + erff(u * 0.70710678118654752f));
                    sM1[(m0 + r) * 256 + n] = f2bf(gel);
                }
            }
            __syncthreads();
            // out += m1c @ W2c^T (accumulators live across chunks)
#pragma unroll
            for (int i = 0; i < 8; ++i) {
                int tile = wv * 8 + i;
                int tm = tile & 3, tn = tile >> 2;
                for (int k0 = 0; k0 < 256; k0 += 32) {
                    __builtin_prefetch(wfc2 + (tn * 16) * 1024 + cc4 * 256 + k0 + 64, 0, 3);
                    bf16x16 a = load_frag_A(sM1 + tm * 16 * 256 + k0, 256);
                    bf16x16 bb = load_frag_B_nk(wfc2 + (tn * 16) * 1024 + cc4 * 256 + k0, 1024);
                    acc2[i] = wmma_bf16(a, bb, acc2[i]);
                }
            }
            __syncthreads();
        }
        // epilogue: + b_fc2 + residual, reverse-window + unshift store
#pragma unroll
        for (int i = 0; i < 8; ++i) {
            int tile = wv * 8 + i;
            int tm = tile & 3, tn = tile >> 2;
            int n = tn * 16 + (lane & 15);
            int m0 = tm * 16 + ((lane >> 4) << 3);
            float b2v = b_fc2[n];
#pragma unroll
            for (int r = 0; r < 8; ++r) {
                int m = m0 + r;
                out[gidx(m, n)] = sX1[m * 256 + n] + acc2[i][r] + b2v;
            }
        }
    }
}

// ---------------- launcher ----------------
extern "C" void kernel_launch(void* const* d_in, const int* in_sizes, int n_in,
                              void* d_out, int out_size, void* d_ws, size_t ws_size,
                              hipStream_t stream) {
    const float* x        = (const float*)d_in[0];
    const float* n1g      = (const float*)d_in[1];
    const float* n1b      = (const float*)d_in[2];
    const float* w_qkv    = (const float*)d_in[3];
    const float* q_bias   = (const float*)d_in[4];
    const float* v_bias   = (const float*)d_in[5];
    const float* ls       = (const float*)d_in[6];
    const float* cpb_w1   = (const float*)d_in[7];
    const float* cpb_b1   = (const float*)d_in[8];
    const float* cpb_w2   = (const float*)d_in[9];
    const float* w_proj   = (const float*)d_in[10];
    const float* b_proj   = (const float*)d_in[11];
    const float* n2g      = (const float*)d_in[12];
    const float* n2b      = (const float*)d_in[13];
    const float* w_fc1    = (const float*)d_in[14];
    const float* b_fc1    = (const float*)d_in[15];
    const float* w_fc2    = (const float*)d_in[16];
    const float* b_fc2    = (const float*)d_in[17];
    float* out            = (float*)d_out;

    char* ws = (char*)d_ws;
    unsigned short* wqkv_bf  = (unsigned short*)(ws + 0);        // 768*256
    unsigned short* wproj_bf = (unsigned short*)(ws + 393216);   // 256*256
    unsigned short* wfc1_bf  = (unsigned short*)(ws + 524288);   // 1024*256
    unsigned short* wfc2_bf  = (unsigned short*)(ws + 1048576);  // 256*1024
    float*          rpb      = (float*)(ws + 1572864);           // 225*8

    conv_bf16_kernel<<<(196608 + 255) / 256, 256, 0, stream>>>(w_qkv, wqkv_bf, 196608);
    conv_bf16_kernel<<<(65536 + 255) / 256, 256, 0, stream>>>(w_proj, wproj_bf, 65536);
    conv_bf16_kernel<<<(262144 + 255) / 256, 256, 0, stream>>>(w_fc1, wfc1_bf, 262144);
    conv_bf16_kernel<<<(262144 + 255) / 256, 256, 0, stream>>>(w_fc2, wfc2_bf, 262144);
    cpb_kernel<<<225, 64, 0, stream>>>(cpb_w1, cpb_b1, cpb_w2, rpb);

    (void)hipFuncSetAttribute((const void*)swin_window_kernel,
                              hipFuncAttributeMaxDynamicSharedMemorySize, POOL_BYTES);
    swin_window_kernel<<<CB * CNW, 256, POOL_BYTES, stream>>>(
        x, n1g, n1b, q_bias, v_bias, ls, b_proj, n2g, n2b, b_fc1, b_fc2,
        wqkv_bf, wproj_bf, wfc1_bf, wfc2_bf, rpb, out);
}
